// LSTMTemporalEncoder_50672024158228
// MI455X (gfx1250) — compile-verified, matrix operand order checked by reading, not run on hardware
//
#include <hip/hip_runtime.h>
#include <hip/hip_bf16.h>

// Problem constants
#define BB   256   // batch
#define TT   512   // seq len
#define HH   256   // hidden
#define GG   1024  // 4*H gates
#define MT   16    // batch rows per workgroup (WMMA M)
#define NWG  (BB / MT)          // 16 workgroups
#define NTILES (GG / 16)        // 64 N tiles of 16
#define KSTEPS (HH / 32)        // 8 K steps of 32
#define WAVES 8                 // 256 threads / wave32
#define FRAG_HALVES 512         // 32 lanes * 16 halves per fragment
#define MAT_HALVES (NTILES * KSTEPS * FRAG_HALVES)  // 262144 per matrix

typedef _Float16 v16h __attribute__((ext_vector_type(16)));
typedef _Float16 v8h  __attribute__((ext_vector_type(8)));
typedef float    v8f  __attribute__((ext_vector_type(8)));

// ---------------------------------------------------------------------------
// K mapping for 16-bit WMMA operands (CDNA5 ISA 7.12.2):
//   lanes 0-15 : K = 0..7 (regs 0-3), 16..23 (regs 4-7)
//   lanes 16-31: K = 8..15,            24..31
__device__ __forceinline__ int k_of(int lane, int j) {
    int base = (j < 8) ? j : (j + 8);
    return base + ((lane >= 16) ? 8 : 0);
}

// ---------------------------------------------------------------------------
// Pack W (4H x H, row major, f32) into f16 B-matrix fragments:
// layout [mat][ntile*KSTEPS+kstep][lane][j], element = W[n, k]
__global__ void pack_weights_kernel(const float* __restrict__ whh0,
                                    const float* __restrict__ wih1,
                                    const float* __restrict__ whh1,
                                    _Float16* __restrict__ pack) {
    int gid = blockIdx.x * blockDim.x + threadIdx.x;
    if (gid >= 3 * MAT_HALVES) return;
    int mat = gid / MAT_HALVES;
    int r   = gid % MAT_HALVES;
    int j    = r & 15;
    int lane = (r >> 4) & 31;
    int frag = r >> 9;            // ntile*KSTEPS + kstep
    int ks = frag % KSTEPS;
    int nt = frag / KSTEPS;
    int n = nt * 16 + (lane & 15);
    int k = 32 * ks + k_of(lane, j);
    const float* W = (mat == 0) ? whh0 : (mat == 1) ? wih1 : whh1;
    pack[gid] = (_Float16)W[n * HH + k];
}

// ---------------------------------------------------------------------------
// Native CDNA5 tanh trans op when available; sigmoid via tanh identity.
#if __has_builtin(__builtin_amdgcn_tanhf)
__device__ __forceinline__ float ftanh(float x) { return __builtin_amdgcn_tanhf(x); }
#else
__device__ __forceinline__ float ftanh(float x) { return tanhf(x); }
#endif
__device__ __forceinline__ float fsig(float x) { return 0.5f + 0.5f * ftanh(0.5f * x); }

// Assemble A fragment (16x32 f16) for row block from LDS h buffer ([MT][HH] f16)
__device__ __forceinline__ v16h load_a_frag(const _Float16* h, int lane, int ks) {
    int m   = lane & 15;
    int off = (lane >= 16) ? 8 : 0;
    const _Float16* base = h + m * HH + 32 * ks + off;
    v8h lo = *reinterpret_cast<const v8h*>(base);
    v8h hi = *reinterpret_cast<const v8h*>(base + 16);
    v16h a;
#pragma unroll
    for (int i = 0; i < 8; ++i) { a[i] = lo[i]; a[i + 8] = hi[i]; }
    return a;
}

__device__ __forceinline__ v16h load_b_frag(const _Float16* packmat, int nt, int ks, int lane) {
    return *reinterpret_cast<const v16h*>(
        packmat + (size_t)(nt * KSTEPS + ks) * FRAG_HALVES + lane * 16);
}

// ---------------------------------------------------------------------------
// Persistent fused 2-layer LSTM. One workgroup owns MT=16 batch rows for all
// T timesteps of both layers. Gate->wave mapping keeps all 4 gates of a unit
// column in one wave's accumulators, so the cell update is fully in-register.
//   wave w, slot s in {0,1}, gate g4 in {0..3}:  ntile = g4*16 + 2*w + s
//   -> unit columns j in [32w, 32w+32)
__global__ void __launch_bounds__(256)
lstm2_kernel(const float* __restrict__ seq,    // (B,T,1)
             const float* __restrict__ wih0,   // (4H,1)
             const float* __restrict__ bih0,
             const float* __restrict__ bhh0,
             const float* __restrict__ bih1,
             const float* __restrict__ bhh1,
             const _Float16* __restrict__ pack, // packed whh0|wih1|whh1
             float* __restrict__ out)           // (B,H)
{
    __shared__ _Float16 s_h0[MT * HH];   // 8 KB
    __shared__ _Float16 s_h1[MT * HH];   // 8 KB
    __shared__ float    s_wih0[GG];      // 4 KB
    __shared__ float    s_b0[GG];        // 4 KB
    __shared__ float    s_b1[GG];        // 4 KB
    __shared__ float    s_seq[MT];

    const int tid  = threadIdx.x;
    const int lane = tid & 31;
    const int wave = tid >> 5;
    const int b0   = blockIdx.x * MT;
    const int mo   = (lane >= 16) ? 8 : 0;
    const int ncol = lane & 15;

    const _Float16* whh0p = pack;
    const _Float16* wih1p = pack + MAT_HALVES;
    const _Float16* whh1p = pack + 2 * MAT_HALVES;

    // ---- init ----
    for (int i = tid; i < MT * HH; i += 256) { s_h0[i] = (_Float16)0.f; s_h1[i] = (_Float16)0.f; }
    for (int i = tid; i < GG; i += 256) {
        s_wih0[i] = wih0[i];
        s_b0[i]   = bih0[i] + bhh0[i];
        s_b1[i]   = bih1[i] + bhh1[i];
    }
    float c0[16], c1[16];     // [s*8+v] cell state, in-register
#pragma unroll
    for (int u = 0; u < 16; ++u) { c0[u] = 0.f; c1[u] = 0.f; }
    __syncthreads();

    for (int t = 0; t < TT; ++t) {
        // Opaque scheduling barrier: forbid hoisting the (t-invariant) weight
        // loads out of the timestep loop -> they stream from L2, not scratch.
        asm volatile("" ::: "memory");

        if (tid < MT) s_seq[tid] = seq[(size_t)(b0 + tid) * TT + t];
        __syncthreads();

        // ======== layer 0 GEMM: gates = h0 @ Whh0^T ========
        v8f acc[4][2];
#pragma unroll
        for (int g4 = 0; g4 < 4; ++g4)
#pragma unroll
            for (int s = 0; s < 2; ++s) acc[g4][s] = (v8f){};

#pragma unroll 1
        for (int ks = 0; ks < KSTEPS; ++ks) {
            v16h a = load_a_frag(s_h0, lane, ks);      // reused by 8 WMMAs
#pragma unroll
            for (int g4 = 0; g4 < 4; ++g4) {
#pragma unroll
                for (int s = 0; s < 2; ++s) {
                    int nt = g4 * 16 + 2 * wave + s;
                    v16h b = load_b_frag(whh0p, nt, ks, lane);
                    acc[g4][s] = __builtin_amdgcn_wmma_f32_16x16x32_f16(
                        false, a, false, b, (short)0, acc[g4][s], false, false);
                }
            }
        }
        __syncthreads();   // all waves done reading s_h0

        // ---- layer 0 cell update, fully in-register ----
        {
            float sm[8];
#pragma unroll
            for (int v = 0; v < 8; ++v) sm[v] = s_seq[v + mo];
#pragma unroll
            for (int s = 0; s < 2; ++s) {
                const int j = 32 * wave + 16 * s + ncol;
                float wi[4], bb[4];
#pragma unroll
                for (int g4 = 0; g4 < 4; ++g4) {
                    wi[g4] = s_wih0[g4 * HH + j];
                    bb[g4] = s_b0[g4 * HH + j];
                }
#pragma unroll
                for (int v = 0; v < 8; ++v) {
                    const int m = v + mo;
                    float xi = acc[0][s][v] + sm[v] * wi[0] + bb[0];
                    float xf = acc[1][s][v] + sm[v] * wi[1] + bb[1];
                    float xg = acc[2][s][v] + sm[v] * wi[2] + bb[2];
                    float xo = acc[3][s][v] + sm[v] * wi[3] + bb[3];
                    float cn = fsig(xf) * c0[s * 8 + v] + fsig(xi) * ftanh(xg);
                    c0[s * 8 + v] = cn;
                    float h = fsig(xo) * ftanh(cn);
                    s_h0[m * HH + j] = (_Float16)h;
                }
            }
        }
        __syncthreads();   // s_h0(t) visible to all waves

        // ======== layer 1 GEMM: gates = h0_t @ Wih1^T + h1 @ Whh1^T ========
#pragma unroll
        for (int g4 = 0; g4 < 4; ++g4)
#pragma unroll
            for (int s = 0; s < 2; ++s) acc[g4][s] = (v8f){};

#pragma unroll 1
        for (int ks = 0; ks < KSTEPS; ++ks) {
            v16h a = load_a_frag(s_h0, lane, ks);
#pragma unroll
            for (int g4 = 0; g4 < 4; ++g4) {
#pragma unroll
                for (int s = 0; s < 2; ++s) {
                    int nt = g4 * 16 + 2 * wave + s;
                    v16h b = load_b_frag(wih1p, nt, ks, lane);
                    acc[g4][s] = __builtin_amdgcn_wmma_f32_16x16x32_f16(
                        false, a, false, b, (short)0, acc[g4][s], false, false);
                }
            }
        }
#pragma unroll 1
        for (int ks = 0; ks < KSTEPS; ++ks) {
            v16h a = load_a_frag(s_h1, lane, ks);
#pragma unroll
            for (int g4 = 0; g4 < 4; ++g4) {
#pragma unroll
                for (int s = 0; s < 2; ++s) {
                    int nt = g4 * 16 + 2 * wave + s;
                    v16h b = load_b_frag(whh1p, nt, ks, lane);
                    acc[g4][s] = __builtin_amdgcn_wmma_f32_16x16x32_f16(
                        false, a, false, b, (short)0, acc[g4][s], false, false);
                }
            }
        }
        __syncthreads();   // all waves done reading s_h0/s_h1

        // ---- layer 1 cell update, fully in-register ----
#pragma unroll
        for (int s = 0; s < 2; ++s) {
            const int j = 32 * wave + 16 * s + ncol;
            float bb[4];
#pragma unroll
            for (int g4 = 0; g4 < 4; ++g4) bb[g4] = s_b1[g4 * HH + j];
#pragma unroll
            for (int v = 0; v < 8; ++v) {
                const int m = v + mo;
                float xi = acc[0][s][v] + bb[0];
                float xf = acc[1][s][v] + bb[1];
                float xg = acc[2][s][v] + bb[2];
                float xo = acc[3][s][v] + bb[3];
                float cn = fsig(xf) * c1[s * 8 + v] + fsig(xi) * ftanh(xg);
                c1[s * 8 + v] = cn;
                float h = fsig(xo) * ftanh(cn);
                s_h1[m * HH + j] = (_Float16)h;
                if (t == TT - 1) out[(size_t)(b0 + m) * HH + j] = h;
            }
        }
        __syncthreads();   // s_h1(t) stable before next step's GEMM
    }
}

// ---------------------------------------------------------------------------
extern "C" void kernel_launch(void* const* d_in, const int* in_sizes, int n_in,
                              void* d_out, int out_size, void* d_ws, size_t ws_size,
                              hipStream_t stream) {
    const float* seq  = (const float*)d_in[0];
    const float* Wih0 = (const float*)d_in[1];
    const float* Whh0 = (const float*)d_in[2];
    const float* bih0 = (const float*)d_in[3];
    const float* bhh0 = (const float*)d_in[4];
    const float* Wih1 = (const float*)d_in[5];
    const float* Whh1 = (const float*)d_in[6];
    const float* bih1 = (const float*)d_in[7];
    const float* bhh1 = (const float*)d_in[8];

    _Float16* pack = (_Float16*)d_ws;   // 3 * 512 KB of packed f16 fragments

    const int total = 3 * MAT_HALVES;
    pack_weights_kernel<<<(total + 255) / 256, 256, 0, stream>>>(Whh0, Wih1, Whh1, pack);

    lstm2_kernel<<<NWG, 256, 0, stream>>>(
        seq, Wih0, bih0, bhh0, bih1, bhh1, pack, (float*)d_out);
    (void)in_sizes; (void)n_in; (void)out_size; (void)ws_size;
}
